// MultiHeadAttention_91311004713550
// MI455X (gfx1250) — compile-verified, matrix operand order checked by reading, not run on hardware
//
#include <hip/hip_runtime.h>

// ---------------------------------------------------------------------------
// MI455X (gfx1250) relative-position multi-head attention, bf16 WMMA pipeline
// ---------------------------------------------------------------------------

typedef __bf16 bf16;
typedef __attribute__((ext_vector_type(16))) __bf16 v16bf;
typedef __attribute__((ext_vector_type(8)))  __bf16 v8bf;
typedef __attribute__((ext_vector_type(2)))  __bf16 v2bf;
typedef __attribute__((ext_vector_type(8)))  float  v8f;
typedef __attribute__((ext_vector_type(4)))  float  v4f;

#define Bdim 8
#define Cdim 512
#define Tdim 1024
#define Hdim 8
#define KCd  64
#define SSTR 1028   // S row stride (floats): 1028 % 64 == 4 -> conflict-reduced

__device__ __forceinline__ v8bf ld8(const bf16* p) {
  return *reinterpret_cast<const v8bf*>(p);
}

__device__ __forceinline__ v16bf mk16(v8bf lo, v8bf hi) {
  v16bf r;
#pragma unroll
  for (int i = 0; i < 8; ++i) { r[i] = lo[i]; r[i + 8] = hi[i]; }
  return r;
}

__device__ __forceinline__ v8f wmma_bf(v16bf a, v16bf b, v8f c) {
  return __builtin_amdgcn_wmma_f32_16x16x32_bf16(false, a, false, b,
                                                 (short)0, c, false, false);
}

__device__ __forceinline__ unsigned pk_bf16(float a, float b) {
  v2bf t; t[0] = (bf16)a; t[1] = (bf16)b;
  return __builtin_bit_cast(unsigned, t);
}

// CDNA5 async global->LDS copy (ASYNCcnt), 16 bytes per lane.
__device__ __forceinline__ void async_g2l_b128(unsigned lds_off, const void* g) {
  asm volatile("global_load_async_to_lds_b128 %0, %1, off"
               :: "v"(lds_off), "v"(g) : "memory");
}
__device__ __forceinline__ void wait_async0() {
  asm volatile("s_wait_asynccnt 0" ::: "memory");
}

// ---------------------------------------------------------------------------
// Kernel 1: transpose + convert weights f32[C][C] -> bf16 Wt[d][c] = W[c][d]
// ---------------------------------------------------------------------------
__global__ void wcvt_kernel(const float* __restrict__ W, bf16* __restrict__ Wt) {
  for (int i = blockIdx.x * 256 + threadIdx.x; i < Cdim * Cdim;
       i += gridDim.x * 256) {
    int d = i >> 9, c = i & 511;
    Wt[i] = (bf16)W[c * Cdim + d];
  }
}

// ---------------------------------------------------------------------------
// Kernel 2: projection  out = (X^T W + bias) * scale
// Block 256 thr = 8 waves; block tile 64t x 128d; wave tile 32x32 (4 WMMA/step)
// vlayout==0: out[b][h][t][kc] (q/k);  vlayout==1: out[b][h][kc][t] (v)
// ---------------------------------------------------------------------------
__global__ __launch_bounds__(256) void proj_kernel(
    const float* __restrict__ in, const bf16* __restrict__ Wt,
    const float* __restrict__ bias, bf16* __restrict__ out,
    float scale, int vlayout) {
  __shared__ alignas(16) unsigned Xt32[64][20];  // [t][c-pair], bf16 stride 40
  const bf16* XtB = reinterpret_cast<const bf16*>(&Xt32[0][0]);

  int idx = blockIdx.x;               // 8b x 16tb x 4db = 512
  int b  = idx >> 6;
  int tb = (idx >> 2) & 15;
  int db = idx & 3;
  int t0 = tb * 64, d0 = db * 128;

  int tid  = threadIdx.x;
  int lane = tid & 31, w = tid >> 5;
  int ln   = lane & 15;
  int hf   = lane >> 4;
  int koff = hf ? 8 : 0;
  int rtw  = w & 1;    // 32-row half
  int dtw  = w >> 1;   // 32-col quarter of 128

  v8f acc[2][2];
#pragma unroll
  for (int r = 0; r < 2; ++r)
#pragma unroll
    for (int d2 = 0; d2 < 2; ++d2)
#pragma unroll
      for (int i = 0; i < 8; ++i) acc[r][d2][i] = 0.f;

  int cp = tid & 15;   // c-pair 0..15 (c = 2cp, 2cp+1)
  int tq = tid >> 4;   // 0..15 (t group of 4)
  const float* gin = in + (size_t)b * Cdim * Tdim;

  for (int ks = 0; ks < Cdim / 32; ++ks) {
    int c0 = ks * 32;
    // stage x[b, c0..c0+31, t0..t0+63] -> Xt[t][c] bf16 (packed b32 stores)
    const float* s0p = gin + (size_t)(c0 + 2 * cp) * Tdim + t0 + tq * 4;
    v4f r0 = *reinterpret_cast<const v4f*>(s0p);
    v4f r1 = *reinterpret_cast<const v4f*>(s0p + Tdim);
#pragma unroll
    for (int i = 0; i < 4; ++i) Xt32[tq * 4 + i][cp] = pk_bf16(r0[i], r1[i]);
    __syncthreads();

    const bf16* a0p = XtB + (rtw * 32 + ln) * 40 + koff;
    const bf16* a1p = a0p + 16 * 40;
    v16bf a0 = mk16(ld8(a0p), ld8(a0p + 16));
    v16bf a1 = mk16(ld8(a1p), ld8(a1p + 16));
    const bf16* wr0 = Wt + (size_t)(d0 + dtw * 32 + ln) * Cdim + c0 + koff;
    const bf16* wr1 = wr0 + (size_t)16 * Cdim;
    __builtin_prefetch(wr0 + 32, 0, 3);
    v16bf b0 = mk16(ld8(wr0), ld8(wr0 + 16));
    v16bf b1 = mk16(ld8(wr1), ld8(wr1 + 16));
    acc[0][0] = wmma_bf(a0, b0, acc[0][0]);
    acc[0][1] = wmma_bf(a0, b1, acc[0][1]);
    acc[1][0] = wmma_bf(a1, b0, acc[1][0]);
    acc[1][1] = wmma_bf(a1, b1, acc[1][1]);
    __syncthreads();
  }

#pragma unroll
  for (int r = 0; r < 2; ++r)
#pragma unroll
    for (int d2 = 0; d2 < 2; ++d2) {
      int d = d0 + dtw * 32 + d2 * 16 + ln;
      float bia = bias[d];
      int hh = d >> 6, kc = d & 63;
#pragma unroll
      for (int v = 0; v < 8; ++v) {
        int t = t0 + rtw * 32 + r * 16 + hf * 8 + v;
        float val = (acc[r][d2][v] + bia) * scale;
        if (!vlayout)
          out[(((size_t)b * Hdim + hh) * Tdim + t) * KCd + kc] = (bf16)val;
        else
          out[(((size_t)b * Hdim + hh) * KCd + kc) * Tdim + t] = (bf16)val;
      }
    }
}

// ---------------------------------------------------------------------------
// Kernel 3: attention for one (b, h, 32-row block).
//   stage: Q rows via async global->LDS (ASYNCcnt), ek/ev bf16 tables
//   pass1: S = Qs K^T (WMMA) + rel-k band (WMMA vs ek, scattered into band)
//   pass2: row max / exp / row sum (unnormalized e kept in LDS f32)
//   pass3: O = (e V + band-WMMA vs ev^T) / rowsum -> bf16 [b][t][h*64+kc]
// ---------------------------------------------------------------------------
__global__ __launch_bounds__(256) void attn_kernel(
    const bf16* __restrict__ qb, const bf16* __restrict__ kb,
    const bf16* __restrict__ vb, const float* __restrict__ ekg,
    const float* __restrict__ evg, bf16* __restrict__ ob) {
  extern __shared__ char smem[];
  constexpr int ROWRED_OFF = 32 * SSTR * 4;        // 131584
  constexpr int ROWSUM_OFF = ROWRED_OFF + 256 * 4; // 132608
  constexpr int QS_OFF     = ROWSUM_OFF + 32 * 4;  // 132736 (stride 72 bf16)
  constexpr int EKB_OFF    = QS_OFF + 32 * 144;    // 137344 (16 x 72 bf16)
  constexpr int EVT_OFF    = EKB_OFF + 16 * 144;   // 139648 (64 x 40 bf16)

  float* Sbuf   = reinterpret_cast<float*>(smem);
  float* rowred = reinterpret_cast<float*>(smem + ROWRED_OFF);
  float* rowsum = reinterpret_cast<float*>(smem + ROWSUM_OFF);
  bf16*  Qs     = reinterpret_cast<bf16*>(smem + QS_OFF);
  bf16*  ekB    = reinterpret_cast<bf16*>(smem + EKB_OFF);
  bf16*  evT    = reinterpret_cast<bf16*>(smem + EVT_OFF);

  int idx = blockIdx.x;
  int b  = idx >> 8;
  int h  = (idx >> 5) & 7;
  int rb = idx & 31;
  int i0 = rb * 32;

  int tid  = threadIdx.x;
  int lane = tid & 31, w = tid >> 5;
  int ln   = lane & 15;
  int hf   = lane >> 4;
  int koff = hf ? 8 : 0;

  size_t bh = (size_t)b * Hdim + h;
  const bf16* qbh = qb + bh * Tdim * KCd;
  const bf16* kbh = kb + bh * Tdim * KCd;
  const bf16* vbh = vb + bh * KCd * Tdim;

  // ---- stage: async-copy Q row block (32 x 64 bf16) into padded LDS ----
  {
    int qr = tid >> 3, qc = tid & 7;  // 32 rows x 8 x 16B chunks
    unsigned laddr = (unsigned)(size_t)(Qs + qr * 72 + qc * 8);
    const bf16* g = qbh + (size_t)(i0 + qr) * KCd + qc * 8;
    async_g2l_b128(laddr, g);
  }
  // ek table as B-matrix [16][72] bf16 (rows 9..15 and cols 64.. zero)
  for (int i = tid; i < 16 * 72; i += 256) {
    int m = i / 72, kc = i % 72;
    ekB[i] = (bf16)((m < 9 && kc < 64) ? ekg[m * 64 + kc] : 0.f);
  }
  // ev^T table [64][40] bf16: evT[kc][d] = ev[d][kc], d>=9 zero
  for (int i = tid; i < 64 * 40; i += 256) {
    int kc = i / 40, d = i % 40;
    evT[i] = (bf16)((d < 9) ? evg[d * 64 + kc] : 0.f);
  }
  wait_async0();
  __syncthreads();

  // ---- pass 1: S = Qs K^T ----
  int rt = w & 1;
  const bf16* qp = Qs + (rt * 16 + ln) * 72;
  v16bf a0 = mk16(ld8(qp + koff),      ld8(qp + koff + 16));
  v16bf a1 = mk16(ld8(qp + 32 + koff), ld8(qp + 32 + koff + 16));
  for (int ct = (w >> 1); ct < 64; ct += 4) {
    int s = ct * 16 + ln;
    const bf16* kp = kbh + (size_t)s * KCd;
    __builtin_prefetch(kp + 64 * KCd, 0, 3);
    v16bf b0 = mk16(ld8(kp + koff),      ld8(kp + koff + 16));
    v16bf b1 = mk16(ld8(kp + 32 + koff), ld8(kp + 32 + koff + 16));
    v8f acc;
#pragma unroll
    for (int z = 0; z < 8; ++z) acc[z] = 0.f;
    acc = wmma_bf(a0, b0, acc);
    acc = wmma_bf(a1, b1, acc);
    int j = ct * 16 + ln;
#pragma unroll
    for (int v = 0; v < 8; ++v) {
      int il = rt * 16 + hf * 8 + v;
      Sbuf[(size_t)il * SSTR + j] = acc[v];
    }
  }
  __syncthreads();

  // ---- rel-k band: Rel[i, m] = q[i] . ek[m]  (one WMMA pair per row tile) ----
  if (w < 2) {
    const bf16* e0 = ekB + ln * 72 + koff;
    v16bf eb0 = mk16(ld8(e0),      ld8(e0 + 16));
    v16bf eb1 = mk16(ld8(e0 + 32), ld8(e0 + 32 + 16));
    v8f r;
#pragma unroll
    for (int z = 0; z < 8; ++z) r[z] = 0.f;
    r = wmma_bf(a0, eb0, r);
    r = wmma_bf(a1, eb1, r);
    if (ln < 9) {
      int d = ln - 4;
#pragma unroll
      for (int v = 0; v < 8; ++v) {
        int il = rt * 16 + hf * 8 + v;
        int j = i0 + il + d;
        if (j >= 0 && j < Tdim) Sbuf[(size_t)il * SSTR + j] += r[v];
      }
    }
  }
  __syncthreads();

  // ---- pass 2: softmax (keep unnormalized exp, save row sums) ----
  {
    int row = tid >> 3, seg = tid & 7;
    float* srow = Sbuf + (size_t)row * SSTR + seg * 128;
    float m = -3.0e38f;
    for (int ci = 0; ci < 128; ++ci) m = fmaxf(m, srow[ci]);
    rowred[row * 8 + seg] = m;
    __syncthreads();
    float rm = rowred[row * 8];
#pragma unroll
    for (int s2 = 1; s2 < 8; ++s2) rm = fmaxf(rm, rowred[row * 8 + s2]);
    __syncthreads();  // all reads of rowred done before rewrite
    float ssum = 0.f;
    for (int ci = 0; ci < 128; ++ci) {
      float e = __expf(srow[ci] - rm);
      srow[ci] = e;
      ssum += e;
    }
    rowred[row * 8 + seg] = ssum;
    __syncthreads();
    if (seg == 0) {
      float rs = 0.f;
#pragma unroll
      for (int s2 = 0; s2 < 8; ++s2) rs += rowred[row * 8 + s2];
      rowsum[row] = rs;
    }
  }
  __syncthreads();

  // ---- pass 3: O = (e V + band(e) ev) / rowsum ----
  {
    int rt3 = w & 1;
    int nt  = w >> 1;  // kc tile 0..3
    v8f acc;
#pragma unroll
    for (int z = 0; z < 8; ++z) acc[z] = 0.f;

    for (int jc = 0; jc < Tdim / 32; ++jc) {
      int row = rt3 * 16 + ln;
      const float* s0 = Sbuf + (size_t)row * SSTR + jc * 32 + koff;
      v4f f0 = *reinterpret_cast<const v4f*>(s0);
      v4f f1 = *reinterpret_cast<const v4f*>(s0 + 4);
      v4f f2 = *reinterpret_cast<const v4f*>(s0 + 16);
      v4f f3 = *reinterpret_cast<const v4f*>(s0 + 20);
      v16bf a;
#pragma unroll
      for (int i = 0; i < 4; ++i) {
        a[i]      = (bf16)f0[i];
        a[4 + i]  = (bf16)f1[i];
        a[8 + i]  = (bf16)f2[i];
        a[12 + i] = (bf16)f3[i];
      }
      int kcn = nt * 16 + ln;
      const bf16* vp = vbh + (size_t)kcn * Tdim + jc * 32 + koff;
      acc = wmma_bf(a, mk16(ld8(vp), ld8(vp + 16)), acc);
    }

    // rel-v band as one extra WMMA: A = gathered band of e, B = ev^T
    {
      int ilr = rt3 * 16 + ln;  // A row for this lane
      v16bf ab;
#pragma unroll
      for (int e = 0; e < 16; ++e) {
        int K = koff + (e < 8 ? e : e + 8);  // fragment K index
        float val = 0.f;
        if (K < 9) {
          int j = i0 + ilr + K - 4;
          if (j >= 0 && j < Tdim) val = Sbuf[(size_t)ilr * SSTR + j];
        }
        ab[e] = (bf16)val;
      }
      const bf16* ev0 = evT + (nt * 16 + ln) * 40 + koff;
      acc = wmma_bf(ab, mk16(ld8(ev0), ld8(ev0 + 16)), acc);
    }

#pragma unroll
    for (int v = 0; v < 8; ++v) {
      int il = rt3 * 16 + hf * 8 + v;
      int ia = i0 + il;
      int kc = nt * 16 + ln;
      float val = acc[v] / rowsum[il];
      ob[((size_t)b * Tdim + ia) * Cdim + h * KCd + kc] = (bf16)val;
    }
  }
}

// ---------------------------------------------------------------------------
// Kernel 4: output projection  y[b][o][t] = sum_c O'[b][t][c] WoT[o][c] + bo
// Block tile 64t x 128o, wave tile 32x32 (4 WMMA/step). f32 b128 stores.
// ---------------------------------------------------------------------------
__global__ __launch_bounds__(256) void projo_kernel(
    const bf16* __restrict__ ob, const bf16* __restrict__ Wot,
    const float* __restrict__ bo, float* __restrict__ y) {
  int idx = blockIdx.x;               // 8b x 16tb x 4ob = 512
  int b  = idx >> 6;
  int tb = (idx >> 2) & 15;
  int od = idx & 3;
  int t0 = tb * 64, o0 = od * 128;

  int tid  = threadIdx.x;
  int lane = tid & 31, w = tid >> 5;
  int ln   = lane & 15;
  int hf   = lane >> 4;
  int koff = hf ? 8 : 0;
  int rtw  = w & 1;
  int dtw  = w >> 1;

  v8f acc[2][2];
#pragma unroll
  for (int r = 0; r < 2; ++r)
#pragma unroll
    for (int d2 = 0; d2 < 2; ++d2)
#pragma unroll
      for (int i = 0; i < 8; ++i) acc[r][d2][i] = 0.f;

  const bf16* ar0 = ob + ((size_t)b * Tdim + t0 + rtw * 32 + ln) * Cdim;
  const bf16* ar1 = ar0 + (size_t)16 * Cdim;
  const bf16* wr0 = Wot + (size_t)(o0 + dtw * 32 + ln) * Cdim;
  const bf16* wr1 = wr0 + (size_t)16 * Cdim;

  for (int cs = 0; cs < Cdim; cs += 32) {
    __builtin_prefetch(ar0 + cs + 128, 0, 3);
    v16bf a0 = mk16(ld8(ar0 + cs + koff), ld8(ar0 + cs + koff + 16));
    v16bf a1 = mk16(ld8(ar1 + cs + koff), ld8(ar1 + cs + koff + 16));
    v16bf b0 = mk16(ld8(wr0 + cs + koff), ld8(wr0 + cs + koff + 16));
    v16bf b1 = mk16(ld8(wr1 + cs + koff), ld8(wr1 + cs + koff + 16));
    acc[0][0] = wmma_bf(a0, b0, acc[0][0]);
    acc[0][1] = wmma_bf(a0, b1, acc[0][1]);
    acc[1][0] = wmma_bf(a1, b0, acc[1][0]);
    acc[1][1] = wmma_bf(a1, b1, acc[1][1]);
  }

#pragma unroll
  for (int r = 0; r < 2; ++r)
#pragma unroll
    for (int d2 = 0; d2 < 2; ++d2) {
      int o = o0 + dtw * 32 + d2 * 16 + ln;
      float bia = bo[o];
      float* dst = y + ((size_t)b * Cdim + o) * Tdim + t0 + rtw * 32 + r * 16 + hf * 8;
      v4f lo, hi;
#pragma unroll
      for (int i = 0; i < 4; ++i) {
        lo[i] = acc[r][d2][i] + bia;
        hi[i] = acc[r][d2][4 + i] + bia;
      }
      *reinterpret_cast<v4f*>(dst)     = lo;
      *reinterpret_cast<v4f*>(dst + 4) = hi;
    }
}

// ---------------------------------------------------------------------------
extern "C" void kernel_launch(void* const* d_in, const int* in_sizes, int n_in,
                              void* d_out, int out_size, void* d_ws,
                              size_t ws_size, hipStream_t stream) {
  (void)in_sizes; (void)n_in; (void)out_size; (void)ws_size;
  const float* x  = (const float*)d_in[0];
  const float* c  = (const float*)d_in[1];
  const float* Wq = (const float*)d_in[2];
  const float* bq = (const float*)d_in[3];
  const float* Wk = (const float*)d_in[4];
  const float* bk = (const float*)d_in[5];
  const float* Wv = (const float*)d_in[6];
  const float* bv = (const float*)d_in[7];
  const float* Wo = (const float*)d_in[8];
  const float* bo = (const float*)d_in[9];
  const float* ek = (const float*)d_in[10];  // [1,9,64]
  const float* ev = (const float*)d_in[11];  // [1,9,64]
  float* y = (float*)d_out;

  char* p = (char*)d_ws;
  auto carve = [&](size_t bytes) -> char* {
    char* r = p;
    p += (bytes + 255) & ~(size_t)255;
    return r;
  };
  const size_t wbytes   = (size_t)Cdim * Cdim * sizeof(bf16);
  const size_t qkvbytes = (size_t)Bdim * Hdim * Tdim * KCd * sizeof(bf16);
  bf16* Wqt = (bf16*)carve(wbytes);
  bf16* Wkt = (bf16*)carve(wbytes);
  bf16* Wvt = (bf16*)carve(wbytes);
  bf16* Wot = (bf16*)carve(wbytes);
  bf16* qb  = (bf16*)carve(qkvbytes);
  bf16* kb  = (bf16*)carve(qkvbytes);
  bf16* vb  = (bf16*)carve(qkvbytes);
  bf16* ob  = (bf16*)carve((size_t)Bdim * Tdim * Cdim * sizeof(bf16));

  wcvt_kernel<<<dim3(256), dim3(256), 0, stream>>>(Wq, Wqt);
  wcvt_kernel<<<dim3(256), dim3(256), 0, stream>>>(Wk, Wkt);
  wcvt_kernel<<<dim3(256), dim3(256), 0, stream>>>(Wv, Wvt);
  wcvt_kernel<<<dim3(256), dim3(256), 0, stream>>>(Wo, Wot);

  const float scale = 0.125f;  // 1/sqrt(64)
  const int proj_blocks = Bdim * (Tdim / 64) * (Cdim / 128);  // 512
  proj_kernel<<<dim3(proj_blocks), dim3(256), 0, stream>>>(x, Wqt, bq, qb, scale, 0);
  proj_kernel<<<dim3(proj_blocks), dim3(256), 0, stream>>>(c, Wkt, bk, kb, 1.0f, 0);
  proj_kernel<<<dim3(proj_blocks), dim3(256), 0, stream>>>(c, Wvt, bv, vb, 1.0f, 1);

  const int attn_blocks = Bdim * Hdim * (Tdim / 32);  // 2048
  const int attn_smem = 32 * SSTR * 4 + 256 * 4 + 32 * 4   // S, rowred, rowsum
                      + 32 * 144 + 16 * 144 + 64 * 80;     // Qs, ekB, evT
  attn_kernel<<<dim3(attn_blocks), dim3(256), attn_smem, stream>>>(qb, kb, vb,
                                                                   ek, ev, ob);

  projo_kernel<<<dim3(proj_blocks), dim3(256), 0, stream>>>(ob, Wot, bo, y);
}
